// HTModel_82265803588345
// MI455X (gfx1250) — matrix operand order repaired and checked by reading, not compiled, not run on hardware
//
#include <hip/hip_runtime.h>

// ---------------------------------------------------------------------------
// MI455X / gfx1250 Hierarchical-Tucker forward pass.
//  * All contractions via v_wmma_f32_16x16x32_bf16 (bf16 in, f32 accum).
//    ~21.7 G MACs => ~2.7M WMMA issues; compute-bound (HBM side is ~5-10us
//    at 23.3 TB/s; all params are L2-resident in the 192MB L2).
//  * Tree/top sibling tiles staged into LDS with the Tensor Data Mover
//    (tensor_load_to_lds + s_wait_tensorcnt), sibling product fused into the
//    WMMA A-fragment load as packed bf16 multiplies.
// ---------------------------------------------------------------------------

typedef __bf16 bf16;
typedef __attribute__((ext_vector_type(16))) __bf16 v16bf;
typedef __attribute__((ext_vector_type(8)))  __bf16 v8bf;
typedef __attribute__((ext_vector_type(8)))  float  v8f;
typedef __attribute__((ext_vector_type(4)))  unsigned int u32x4;
typedef __attribute__((ext_vector_type(8)))  unsigned int u32x8;

#define DEVINL __device__ __forceinline__

static DEVINL float bf2f(bf16 h) { return (float)h; }
static DEVINL bf16  f2bf(float f) { return (bf16)f; }

static DEVINL v8f wmma_bf16(v16bf a, v16bf b, v8f c) {
    // (neg_a, A, neg_b, B, c_mod, C, reuse_a, reuse_b)
    return __builtin_amdgcn_wmma_f32_16x16x32_bf16(false, a, false, b,
                                                   (short)0, c, false, false);
}

// ---------------------------------------------------------------------------
// Tensor Data Mover: 1-D tile (elems bf16, contiguous) from global -> LDS.
// D# per CDNA5 ISA:
//   group0: count=1 | lds_addr[63:32] | global_addr[120:64] | type=2[127:126]
//   group1: data_size=1 (2B) [17:16] | tensor_dim0[79:48] | tensor_dim1[111:80]
//           | tile_dim0[127:112] | tensor_dim0_stride[207:160]
// Issued by one wave per block; completion via s_wait_tensorcnt.
// ---------------------------------------------------------------------------
static DEVINL void tdm_load_1d(unsigned ldsOff, const void* gptr, unsigned elems) {
    unsigned long long ga = (unsigned long long)(size_t)gptr;
    u32x4 g0;
    g0[0] = 1u;                                              // count=1 (user D#)
    g0[1] = ldsOff;                                          // lds_addr
    g0[2] = (unsigned)ga;                                    // global_addr[31:0]
    g0[3] = (unsigned)((ga >> 32) & 0x1FFFFFFu) | (2u << 30);// addr[56:32] | type=2
    u32x8 g1;
    g1[0] = 1u << 16;                                        // data_size=1 (2 bytes)
    g1[1] = (elems & 0xFFFFu) << 16;                         // tensor_dim0[15:0]
    g1[2] = (elems >> 16) | (1u << 16);                      // tdim0[31:16] | tensor_dim1=1
    g1[3] = (elems & 0xFFFFu) << 16;                         // tile_dim0 = elems
    g1[4] = 0u;                                              // tile_dim1/2 unused
    g1[5] = elems;                                           // tensor_dim0_stride
    g1[6] = 0u;
    g1[7] = 0u;
    asm volatile("tensor_load_to_lds %0, %1" :: "s"(g0), "s"(g1) : "memory");
}

static DEVINL unsigned lds_offset_of(const void* p) {
    // ISA: flat LDS aperture addresses map to LDS via addr[31:0].
    return (unsigned)(size_t)p;
}

// ---- A-fragment (16x32 bf16, M rows per-lane) from row-major LDS tile -----
// lanes 0-15: M=lane, chunks K=[k0+0..7],[k0+16..23]; lanes 16-31: +8.
static DEVINL v16bf load_a_frag(const bf16* base, int ld, int k0, int lane) {
    int m  = lane & 15;
    int hi = (lane >> 4) & 1;
    const bf16* p = base + (size_t)m * ld + k0 + hi * 8;
    v8bf c0 = *(const v8bf*)(p);
    v8bf c1 = *(const v8bf*)(p + 16);
    return __builtin_shufflevector(c0, c1, 0,1,2,3,4,5,6,7,8,9,10,11,12,13,14,15);
}

// ---- A-fragment with fused sibling product: A = L .* R (packed bf16 mul) --
static DEVINL v16bf load_a_frag_prod(const bf16* L, const bf16* R,
                                     int ld, int k0, int lane) {
    int m  = lane & 15;
    int hi = (lane >> 4) & 1;
    size_t o = (size_t)m * ld + k0 + hi * 8;
    v8bf l0 = *(const v8bf*)(L + o);
    v8bf l1 = *(const v8bf*)(L + o + 16);
    v8bf r0 = *(const v8bf*)(R + o);
    v8bf r1 = *(const v8bf*)(R + o + 16);
    v8bf p0 = l0 * r0;                     // v_pk_mul_bf16
    v8bf p1 = l1 * r1;
    return __builtin_shufflevector(p0, p1, 0,1,2,3,4,5,6,7,8,9,10,11,12,13,14,15);
}

// ---- B-fragment (32x16): W row-major (out x in); B[k][n] = W[n0+n][k0+k] --
static DEVINL v16bf load_b_frag(const bf16* W, int Kdim, int n0, int k0, int lane) {
    int n  = lane & 15;
    int hi = (lane >> 4) & 1;
    const bf16* p = W + (size_t)(n0 + n) * Kdim + k0 + hi * 8;
    v8bf c0 = *(const v8bf*)(p);
    v8bf c1 = *(const v8bf*)(p + 16);
    return __builtin_shufflevector(c0, c1, 0,1,2,3,4,5,6,7,8,9,10,11,12,13,14,15);
}

static DEVINL v16bf load_b_frag_clamp(const bf16* W, int Kdim, int n0, int k0,
                                      int lane, int nmax) {
    int n  = lane & 15;
    int hi = (lane >> 4) & 1;
    int row = n0 + n; if (row >= nmax) row = nmax - 1;
    const bf16* p = W + (size_t)row * Kdim + k0 + hi * 8;
    v8bf c0 = *(const v8bf*)(p);
    v8bf c1 = *(const v8bf*)(p + 16);
    return __builtin_shufflevector(c0, c1, 0,1,2,3,4,5,6,7,8,9,10,11,12,13,14,15);
}

// ---- C tile (16x16 f32: VGPR r -> row r+8*hi, col lane&15) -> bf16 LDS ----
static DEVINL void store_c_lds(bf16* out, int ld, int n0, v8f c,
                               const float* bias, bool do_relu, int lane) {
    int n  = lane & 15;
    int hi = (lane >> 4) & 1;
    float bv = bias ? bias[n0 + n] : 0.f;
#pragma unroll
    for (int r = 0; r < 8; ++r) {
        float v = c[r] + bv;
        if (do_relu) v = fmaxf(v, 0.f);
        out[(size_t)(r + hi * 8) * ld + n0 + n] = f2bf(v);
    }
}

// ===========================================================================
// Param packing: f32 -> bf16 (+ transpose for the MLP weights)
// ===========================================================================
__global__ void k_cast(const float* __restrict__ in, bf16* __restrict__ out, int n) {
    for (int i = blockIdx.x * blockDim.x + threadIdx.x; i < n;
         i += gridDim.x * blockDim.x)
        out[i] = f2bf(in[i]);
}

__global__ void k_cast_tr(const float* __restrict__ in, bf16* __restrict__ outT,
                          int ind, int outd) {
    int n = ind * outd;
    for (int idx = blockIdx.x * blockDim.x + threadIdx.x; idx < n;
         idx += gridDim.x * blockDim.x) {
        int o = idx / ind;
        int i = idx - o * ind;
        outT[idx] = f2bf(in[(size_t)i * outd + o]);
    }
}

// ===========================================================================
// Fused representation MLP + leaf contraction (per wave: 16 batch rows at
// fixed leaf j; X->h1(128)->h2(64)->h3(32)->F(32)->t0(64) via LDS staging).
// ===========================================================================
__global__ void __launch_bounds__(128)
k_mlp_leaf(const float* __restrict__ X,
           const bf16* __restrict__ W1t, const float* __restrict__ b1,
           const bf16* __restrict__ W2t, const float* __restrict__ b2,
           const bf16* __restrict__ W3t, const float* __restrict__ b3,
           const bf16* __restrict__ W4t, const float* __restrict__ b4,
           const bf16* __restrict__ P0,
           bf16* __restrict__ t0) {
    __shared__ __align__(16) bf16 sX[4][16 * 64];
    __shared__ __align__(16) bf16 sA[4][16 * 128];
    __shared__ __align__(16) bf16 sB[4][16 * 128];

    const int w    = threadIdx.x >> 5;
    const int lane = threadIdx.x & 31;
    const int j    = blockIdx.y;
    const int b0   = (blockIdx.x * 4 + w) * 16;

#pragma unroll
    for (int i = 0; i < 32; ++i) {
        int e = i * 32 + lane;
        int rr = e >> 6, c = e & 63;
        sX[w][e] = f2bf(X[((size_t)(b0 + rr) * 64 + j) * 64 + c]);
    }
    __syncthreads();

    // h1 = relu(X @ W1 + b1): K=64, N=128
#pragma unroll
    for (int nt = 0; nt < 8; ++nt) {
        v8f acc = {};
#pragma unroll
        for (int kb = 0; kb < 2; ++kb) {
            v16bf a = load_a_frag(sX[w], 64, kb * 32, lane);
            v16bf b = load_b_frag(W1t, 64, nt * 16, kb * 32, lane);
            acc = wmma_bf16(a, b, acc);
        }
        store_c_lds(sA[w], 128, nt * 16, acc, b1, true, lane);
    }
    __syncthreads();

    // h2 = relu(h1 @ W2 + b2): K=128, N=64
#pragma unroll
    for (int nt = 0; nt < 4; ++nt) {
        v8f acc = {};
#pragma unroll
        for (int kb = 0; kb < 4; ++kb) {
            v16bf a = load_a_frag(sA[w], 128, kb * 32, lane);
            v16bf b = load_b_frag(W2t, 128, nt * 16, kb * 32, lane);
            acc = wmma_bf16(a, b, acc);
        }
        store_c_lds(sB[w], 64, nt * 16, acc, b2, true, lane);
    }
    __syncthreads();

    // h3 = relu(h2 @ W3 + b3): K=64, N=32
#pragma unroll
    for (int nt = 0; nt < 2; ++nt) {
        v8f acc = {};
#pragma unroll
        for (int kb = 0; kb < 2; ++kb) {
            v16bf a = load_a_frag(sB[w], 64, kb * 32, lane);
            v16bf b = load_b_frag(W3t, 64, nt * 16, kb * 32, lane);
            acc = wmma_bf16(a, b, acc);
        }
        store_c_lds(sA[w], 32, nt * 16, acc, b3, true, lane);
    }
    __syncthreads();

    // F = h3 @ W4 + b4: K=32, N=32
#pragma unroll
    for (int nt = 0; nt < 2; ++nt) {
        v16bf a = load_a_frag(sA[w], 32, 0, lane);
        v16bf b = load_b_frag(W4t, 32, nt * 16, 0, lane);
        v8f acc = {};
        acc = wmma_bf16(a, b, acc);
        store_c_lds(sB[w], 32, nt * 16, acc, b4, false, lane);
    }
    __syncthreads();

    // Leaf: t0[b, a] = sum_m F[b,m] * P0[j][a][m]  (K=32, N=64)
    const bf16* P0j = P0 + (size_t)j * 64 * 32;
    const int n = lane & 15, hi = (lane >> 4) & 1;
#pragma unroll
    for (int nt = 0; nt < 4; ++nt) {
        v16bf a = load_a_frag(sB[w], 32, 0, lane);
        v16bf b = load_b_frag(P0j, 32, nt * 16, 0, lane);
        v8f acc = {};
        acc = wmma_bf16(a, b, acc);
#pragma unroll
        for (int r = 0; r < 8; ++r)
            t0[((size_t)j * 4096 + b0 + r + hi * 8) * 64 + nt * 16 + n] = f2bf(acc[r]);
    }
}

// ===========================================================================
// One HT tree level. TDM stages both contiguous sibling tiles (16 x rp bf16)
// into LDS; the sibling product is fused into the A-fragment load; 8 waves
// split the output columns. Grid (B/16, nodes, ro/128).
// ===========================================================================
__global__ void __launch_bounds__(256)
k_tree_level(const bf16* __restrict__ tin, bf16* __restrict__ tout,
             const bf16* __restrict__ P, int rp, int ro) {
    __shared__ __align__(16) bf16 sL[16 * 512];
    __shared__ __align__(16) bf16 sR[16 * 512];

    const int j   = blockIdx.y;
    const int b0  = blockIdx.x * 16;
    const int tid = threadIdx.x;

    const bf16* tL = tin + ((size_t)(2 * j)     * 4096 + b0) * rp;
    const bf16* tR = tin + ((size_t)(2 * j + 1) * 4096 + b0) * rp;
    const unsigned elems = 16u * (unsigned)rp;   // contiguous 1-D tile

    if (tid < 32) {                              // wave 0 drives the TDM
        tdm_load_1d(lds_offset_of(sL), tL, elems);
        tdm_load_1d(lds_offset_of(sR), tR, elems);
        __builtin_amdgcn_s_wait_tensorcnt(0);
    }
    __syncthreads();

    const int w = tid >> 5, lane = tid & 31;
    const int n0 = blockIdx.z * 128 + w * 16;
    const bf16* Pj = P + (size_t)j * ro * rp;

    v8f acc = {};
    for (int kb = 0; kb < rp; kb += 32) {
        if (kb + 32 < rp)  // keep the L2-resident weight stream warm
            __builtin_prefetch(Pj + (size_t)(n0 + (lane & 15)) * rp + kb + 32, 0, 3);
        v16bf a = load_a_frag_prod(sL, sR, rp, kb, lane);
        v16bf b = load_b_frag(Pj, rp, n0, kb, lane);
        acc = wmma_bf16(a, b, acc);
    }

    const int n = lane & 15, hi = (lane >> 4) & 1;
#pragma unroll
    for (int r = 0; r < 8; ++r)
        tout[((size_t)j * 4096 + b0 + r + hi * 8) * ro + n0 + n] = f2bf(acc[r]);
}

// ===========================================================================
// Top: pair = t5[0]*t5[1]; out = pair @ Ptop^T (512 -> 1000), TDM-staged.
// Grid (B/16, 8) covers 1024 columns; stores guarded at Y=1000.
// ===========================================================================
__global__ void __launch_bounds__(256)
k_top(const bf16* __restrict__ t5, const bf16* __restrict__ Ptop,
      float* __restrict__ out) {
    __shared__ __align__(16) bf16 sL[16 * 512];
    __shared__ __align__(16) bf16 sR[16 * 512];

    const int b0  = blockIdx.x * 16;
    const int tid = threadIdx.x;

    if (tid < 32) {
        tdm_load_1d(lds_offset_of(sL), t5 + (size_t)b0 * 512, 16 * 512);
        tdm_load_1d(lds_offset_of(sR), t5 + ((size_t)4096 + b0) * 512, 16 * 512);
        __builtin_amdgcn_s_wait_tensorcnt(0);
    }
    __syncthreads();

    const int w = tid >> 5, lane = tid & 31;
    const int n0 = blockIdx.y * 128 + w * 16;

    v8f acc = {};
#pragma unroll 4
    for (int kb = 0; kb < 512; kb += 32) {
        v16bf a = load_a_frag_prod(sL, sR, 512, kb, lane);
        v16bf b = load_b_frag_clamp(Ptop, 512, n0, kb, lane, 1000);
        acc = wmma_bf16(a, b, acc);
    }

    const int n = lane & 15, hi = (lane >> 4) & 1;
    const int y = n0 + n;
    if (y < 1000) {
#pragma unroll
        for (int r = 0; r < 8; ++r)
            out[(size_t)(b0 + r + hi * 8) * 1000 + y] = acc[r];
    }
}

// ===========================================================================
// Workspace layout (bf16 elements): params (~8.1MB) + 2 ping-pong activation
// buffers (33.5MB each) => ~75.3MB.
// ===========================================================================
static const size_t OFF_W1  = 0;                          // 128x64
static const size_t OFF_W2  = OFF_W1  + 128 * 64;         // 64x128
static const size_t OFF_W3  = OFF_W2  + 64 * 128;         // 32x64
static const size_t OFF_W4  = OFF_W3  + 32 * 64;          // 32x32
static const size_t OFF_P0  = OFF_W4  + 32 * 32;          // 64x64x32
static const size_t OFF_P1  = OFF_P0  + 64 * 64 * 32;     // 32x128x64
static const size_t OFF_P2  = OFF_P1  + 32 * 128 * 64;    // 16x256x128
static const size_t OFF_P3  = OFF_P2  + 16 * 256 * 128;   // 8x512x256
static const size_t OFF_P4  = OFF_P3  + 8 * 512 * 256;    // 4x512x512
static const size_t OFF_P5  = OFF_P4  + 4 * 512 * 512;    // 2x512x512
static const size_t OFF_PT  = OFF_P5  + 2 * 512 * 512;    // 1000x512
static const size_t OFF_BUFA = OFF_PT + 1000 * 512;
static const size_t OFF_BUFB = OFF_BUFA + (size_t)64 * 4096 * 64;

extern "C" void kernel_launch(void* const* d_in, const int* in_sizes, int n_in,
                              void* d_out, int out_size, void* d_ws, size_t ws_size,
                              hipStream_t stream) {
    const float* X    = (const float*)d_in[0];
    const float* W1   = (const float*)d_in[1];
    const float* b1   = (const float*)d_in[2];
    const float* W2   = (const float*)d_in[3];
    const float* b2   = (const float*)d_in[4];
    const float* W3   = (const float*)d_in[5];
    const float* b3   = (const float*)d_in[6];
    const float* W4   = (const float*)d_in[7];
    const float* b4   = (const float*)d_in[8];
    const float* P0   = (const float*)d_in[9];
    const float* P1   = (const float*)d_in[10];
    const float* P2   = (const float*)d_in[11];
    const float* P3   = (const float*)d_in[12];
    const float* P4   = (const float*)d_in[13];
    const float* P5   = (const float*)d_in[14];
    const float* Ptop = (const float*)d_in[15];

    bf16* ws = (bf16*)d_ws;
    bf16 *W1t = ws + OFF_W1, *W2t = ws + OFF_W2, *W3t = ws + OFF_W3, *W4t = ws + OFF_W4;
    bf16 *P0p = ws + OFF_P0, *P1p = ws + OFF_P1, *P2p = ws + OFF_P2;
    bf16 *P3p = ws + OFF_P3, *P4p = ws + OFF_P4, *P5p = ws + OFF_P5;
    bf16 *PTp = ws + OFF_PT;
    bf16 *bufA = ws + OFF_BUFA, *bufB = ws + OFF_BUFB;

    auto blocks = [](size_t n) { return (unsigned)((n + 255) / 256); };

    k_cast_tr<<<blocks(128 * 64),  256, 0, stream>>>(W1, W1t, 64, 128);
    k_cast_tr<<<blocks(64 * 128),  256, 0, stream>>>(W2, W2t, 128, 64);
    k_cast_tr<<<blocks(32 * 64),   256, 0, stream>>>(W3, W3t, 64, 32);
    k_cast_tr<<<blocks(32 * 32),   256, 0, stream>>>(W4, W4t, 32, 32);
    k_cast<<<blocks(64 * 64 * 32),   256, 0, stream>>>(P0, P0p, 64 * 64 * 32);
    k_cast<<<blocks(32 * 128 * 64),  256, 0, stream>>>(P1, P1p, 32 * 128 * 64);
    k_cast<<<blocks(16 * 256 * 128), 256, 0, stream>>>(P2, P2p, 16 * 256 * 128);
    k_cast<<<blocks(8 * 512 * 256),  256, 0, stream>>>(P3, P3p, 8 * 512 * 256);
    k_cast<<<blocks(4 * 512 * 512),  256, 0, stream>>>(P4, P4p, 4 * 512 * 512);
    k_cast<<<blocks(2 * 512 * 512),  256, 0, stream>>>(P5, P5p, 2 * 512 * 512);
    k_cast<<<blocks(1000 * 512),     256, 0, stream>>>(Ptop, PTp, 1000 * 512);

    k_mlp_leaf<<<dim3(64, 64), 128, 0, stream>>>(X, W1t, b1, W2t, b2, W3t, b3,
                                                 W4t, b4, P0p, bufA);

    k_tree_level<<<dim3(256, 32, 1), 256, 0, stream>>>(bufA, bufB, P1p,  64, 128);
    k_tree_level<<<dim3(256, 16, 2), 256, 0, stream>>>(bufB, bufA, P2p, 128, 256);
    k_tree_level<<<dim3(256,  8, 4), 256, 0, stream>>>(bufA, bufB, P3p, 256, 512);
    k_tree_level<<<dim3(256,  4, 4), 256, 0, stream>>>(bufB, bufA, P4p, 512, 512);
    k_tree_level<<<dim3(256,  2, 4), 256, 0, stream>>>(bufA, bufB, P5p, 512, 512);

    k_top<<<dim3(256, 8), 256, 0, stream>>>(bufB, PTp, (float*)d_out);
}
